// UncertaintyRefinementAttention_75067438399514
// MI455X (gfx1250) — matrix operand order repaired and checked by reading, not compiled, not run on hardware
//
#include <hip/hip_runtime.h>

typedef __attribute__((ext_vector_type(16))) _Float16 v16h;
typedef __attribute__((ext_vector_type(8)))  _Float16 v8h;
typedef __attribute__((ext_vector_type(8)))  float    v8f;

#define DD   64
#define SSW  48
#define HHW  192
#define NN   2304      // 48*48 tokens per window
#define NWIN 16
#define HW2  36864     // 192*192
#define NEGV (-1e10f)

// ---------------- WMMA helpers (CDNA5 gfx1250, wave32) ----------------

__device__ __forceinline__ v8f wmma16(v16h a, v16h b, v8f c) {
  // D = A(16x32 f16) * B(32x16 f16) + C(16x16 f32)
  return __builtin_amdgcn_wmma_f32_16x16x32_f16(false, a, false, b, (short)0, c,
                                                false, false);
}

// A fragment (16x32): lane m = l&15, g = l>>4; element e -> K = (e&8)*2 + g*8 + (e&7)
// Source: row-major [m][k], 16B-aligned rows. Two b128 loads.
__device__ __forceinline__ v16h ldA16(const _Float16* p, int g) {
  v8h lo = *(const v8h*)(p + g * 8);
  v8h hi = *(const v8h*)(p + 16 + g * 8);
  return __builtin_shufflevector(lo, hi, 0, 1, 2, 3, 4, 5, 6, 7,
                                 8, 9, 10, 11, 12, 13, 14, 15);
}

// B fragment (32x16): lane n = l&15, g = l>>4; element e -> K = (e&8)*2 + g*8 + (e&7)
// Source: row-major [n][k] (i.e. B^T). Two b128 loads.
__device__ __forceinline__ v16h ldB16(const _Float16* p, int g) {
  v8h lo = *(const v8h*)(p + g * 8);
  v8h hi = *(const v8h*)(p + 16 + g * 8);
  return __builtin_shufflevector(lo, hi, 0, 1, 2, 3, 4, 5, 6, 7,
                                 8, 9, 10, 11, 12, 13, 14, 15);
}

// Async copy of 16 bytes global -> LDS (CDNA5 Tensor/async path, ASYNCcnt).
// VDST = LDS byte-offset VGPR, VADDR = 64-bit global address (GV mode).
__device__ __forceinline__ void async_b128(const _Float16* lds_dst,
                                           const _Float16* gsrc) {
  unsigned lds = (unsigned)(uintptr_t)lds_dst;            // AS3 offset in low 32b
  unsigned long long ga = (unsigned long long)(uintptr_t)gsrc;
  asm volatile("global_load_async_to_lds_b128 %0, %1, off" ::"v"(lds), "v"(ga)
               : "memory");
}
__device__ __forceinline__ void wait_async0() {
  asm volatile("s_wait_asynccnt 0x0" ::: "memory");
}

// ---------------- prep kernels ----------------

// Convert Wq/Wk/Wv to f16; build fused Wc = W1 @ Wo (64x64) and bc = b1 + W1 @ bo.
__global__ __launch_bounds__(256) void prep_w_kernel(
    const float* __restrict__ ipw, const float* __restrict__ opw,
    const float* __restrict__ opb, const float* __restrict__ w1,
    const float* __restrict__ b1, _Float16* __restrict__ WqH,
    _Float16* __restrict__ WkH, _Float16* __restrict__ WvH,
    _Float16* __restrict__ WcH, float* __restrict__ bc) {
  int i = blockIdx.x * 256 + threadIdx.x;  // 0..4095
  WqH[i] = (_Float16)ipw[i];
  WkH[i] = (_Float16)ipw[4096 + i];
  WvH[i] = (_Float16)ipw[8192 + i];
  int j = i >> 6, col = i & 63;
  float s = 0.f;
  for (int a = 0; a < 64; ++a) s = fmaf(w1[j * 64 + a], opw[a * 64 + col], s);
  WcH[i] = (_Float16)s;
  if (col == 0) {
    float t = b1[j];
    for (int a = 0; a < 64; ++a) t = fmaf(w1[j * 64 + a], opb[a], t);
    bc[j] = t;
  }
}

// Binarized uncertainty per (window, token)
__global__ __launch_bounds__(256) void prep_u_kernel(
    const float* __restrict__ unc, float* __restrict__ Ubin) {
  int i = blockIdx.x * 256 + threadIdx.x;  // 0..36863
  int w = i / NN, t = i % NN;
  int hh = (w >> 2) * SSW + t / SSW;
  int ww = (w & 3) * SSW + t % SSW;
  Ubin[i] = (unc[hh * HHW + ww] > 0.01f) ? 1.0f : 0.0f;
}

// ---------------- QKV projection (WMMA) ----------------
// grid (36, 16): blockIdx.x = 64-token tile, blockIdx.y = window. 256 thr = 8 waves.
__global__ __launch_bounds__(256) void qkv_proj_kernel(
    const float* __restrict__ x, const float* __restrict__ ref,
    const float* __restrict__ ipb, const _Float16* __restrict__ WqH,
    const _Float16* __restrict__ WkH, const _Float16* __restrict__ WvH,
    _Float16* __restrict__ Qh, _Float16* __restrict__ Kh,
    _Float16* __restrict__ Vt) {
  __shared__ __align__(16) _Float16 xs[64 * 72];
  __shared__ __align__(16) _Float16 rs[64 * 72];
  const int win = blockIdx.y;
  const int t0 = blockIdx.x * 64;
  const int tid = threadIdx.x;

  // Stage x/ref window tile -> LDS [token][chan] f16 (stride 72 halves)
  {
    int c = tid >> 2, q4 = tid & 3;
    #pragma unroll 4
    for (int j = 0; j < 16; ++j) {
      int t = q4 * 16 + j;
      int tg = t0 + t;
      int hh = (win >> 2) * SSW + tg / SSW;
      int ww = (win & 3) * SSW + tg % SSW;
      size_t gi = (size_t)c * HW2 + hh * HHW + ww;
      xs[t * 72 + c] = (_Float16)x[gi];
      rs[t * 72 + c] = (_Float16)ref[gi];
    }
  }
  __syncthreads();

  const int wave = tid >> 5, lane = tid & 31;
  const int g = lane >> 4, hn = lane & 15;
  const int slice = wave & 3;
  const int tloc = slice * 16 + hn;
  const _Float16* src = (wave < 4) ? xs : rs;
  v16h a0 = ldA16(src + tloc * 72, g);
  v16h a1 = ldA16(src + tloc * 72 + 32, g);

  if (wave < 4) {  // Q
    #pragma unroll
    for (int dt = 0; dt < 4; ++dt) {
      int o = dt * 16 + hn;
      v16h b0 = ldB16(WqH + o * 64, g);
      v16h b1 = ldB16(WqH + o * 64 + 32, g);
      float bb = ipb[o];
      v8f acc = {bb, bb, bb, bb, bb, bb, bb, bb};
      acc = wmma16(a0, b0, acc);
      acc = wmma16(a1, b1, acc);
      #pragma unroll
      for (int r = 0; r < 8; ++r) {
        int t = t0 + slice * 16 + r + 8 * g;
        Qh[((size_t)win * NN + t) * DD + o] = (_Float16)acc[r];
      }
    }
  } else {  // K (row-major) and V (transposed [d][t])
    #pragma unroll
    for (int dt = 0; dt < 4; ++dt) {
      int o = dt * 16 + hn;
      v16h b0 = ldB16(WkH + o * 64, g);
      v16h b1 = ldB16(WkH + o * 64 + 32, g);
      float bb = ipb[64 + o];
      v8f acc = {bb, bb, bb, bb, bb, bb, bb, bb};
      acc = wmma16(a0, b0, acc);
      acc = wmma16(a1, b1, acc);
      #pragma unroll
      for (int r = 0; r < 8; ++r) {
        int t = t0 + slice * 16 + r + 8 * g;
        Kh[((size_t)win * NN + t) * DD + o] = (_Float16)acc[r];
      }
    }
    #pragma unroll
    for (int dt = 0; dt < 4; ++dt) {
      int o = dt * 16 + hn;
      v16h b0 = ldB16(WvH + o * 64, g);
      v16h b1 = ldB16(WvH + o * 64 + 32, g);
      float bb = ipb[128 + o];
      v8f acc = {bb, bb, bb, bb, bb, bb, bb, bb};
      acc = wmma16(a0, b0, acc);
      acc = wmma16(a1, b1, acc);
      #pragma unroll
      for (int r = 0; r < 8; ++r) {
        int t = t0 + slice * 16 + r + 8 * g;
        Vt[((size_t)win * DD + o) * NN + t] = (_Float16)acc[r];
      }
    }
  }
}

// ---------------- masked flash attention + fused (W1@Wo) proj + residual ----------------
// grid (36, 16): blockIdx.x = 64-row block, blockIdx.y = window. 128 thr = 4 waves,
// each wave owns 16 query rows. KV tiles are staged ONCE per workgroup into LDS
// with double-buffered async global->LDS copies (ASYNCcnt), shared by all 4 waves.
__global__ __launch_bounds__(128) void flash_kernel(
    const _Float16* __restrict__ Qh, const _Float16* __restrict__ Kh,
    const _Float16* __restrict__ Vt, const float* __restrict__ Ubin,
    const _Float16* __restrict__ WcH, const float* __restrict__ bc,
    const float* __restrict__ x, float* __restrict__ refined) {
  __shared__ __align__(16) _Float16 kbuf[2][32 * 72];  // K tile: 32 tok x 64ch (pad 72)
  __shared__ __align__(16) _Float16 vbuf[2][64 * 40];  // V tile: 64 d x 32 tok (pad 40)
  __shared__ __align__(16) _Float16 p_lds[4][16 * 40];
  __shared__ __align__(16) _Float16 o_lds[4][16 * 72];

  const int win = blockIdx.y;
  const int tid = threadIdx.x;
  const int wave = tid >> 5, lane = tid & 31;
  const int g = lane >> 4, hn = lane & 15;
  const int r0 = blockIdx.x * 64 + wave * 16;

  const _Float16* Qw = Qh + (size_t)win * NN * DD;
  const _Float16* Kw = Kh + (size_t)win * NN * DD;
  const _Float16* Vw = Vt + (size_t)win * DD * NN;
  const float* Uw = Ubin + win * NN;

  // Issue async copies for KV tile starting at kt into buffer `buf`.
  // 128 threads x 4 b128 each: K tile 4KB + V tile 4KB.
  auto stage_kv = [&](int kt, int buf) {
    #pragma unroll
    for (int s = 0; s < 2; ++s) {
      int c = tid + s * 128;          // 0..255
      int j = c >> 3, q = c & 7;      // K: token j, 16B chunk q
      async_b128(&kbuf[buf][j * 72 + q * 8],
                 Kw + (size_t)(kt + j) * DD + q * 8);
    }
    #pragma unroll
    for (int s = 0; s < 2; ++s) {
      int c = tid + s * 128;
      int d = c >> 2, q = c & 3;      // V: dim d, 16B chunk q
      async_b128(&vbuf[buf][d * 40 + q * 8],
                 Vw + (size_t)d * NN + kt + q * 8);
    }
  };

  // Q A-fragments for this wave's 16 rows (kept in registers for whole pass)
  const int tq = r0 + hn;
  v16h aq0 = ldA16(Qw + (size_t)tq * DD, g);
  v16h aq1 = ldA16(Qw + (size_t)tq * DD + 32, g);

  float ur[8], rm[8], rl[8];
  #pragma unroll
  for (int r = 0; r < 8; ++r) {
    ur[r] = Uw[r0 + r + 8 * g];
    rm[r] = -3e38f;
    rl[r] = 0.f;
  }
  const v8f vzero = {0, 0, 0, 0, 0, 0, 0, 0};
  v8f accO[4];
  #pragma unroll
  for (int d = 0; d < 4; ++d) accO[d] = vzero;

  _Float16* pw = &p_lds[wave][0];

  stage_kv(0, 0);  // prologue: tile 0 in flight

  for (int it = 0; it < NN / 32; ++it) {
    const int kt = it * 32;
    const int cur = it & 1;

    wait_async0();    // my async copies into kbuf/vbuf[cur] complete
    __syncthreads();  // ...and everyone else's; prior reads of other buf done
    if (kt + 32 < NN) stage_kv(kt + 32, cur ^ 1);  // overlap next tile copy

    const _Float16* kb = &kbuf[cur][0];
    const _Float16* vb = &vbuf[cur][0];

    const int tk0 = kt + hn, tk1 = kt + 16 + hn;
    v16h bk00 = ldB16(kb + hn * 72, g);
    v16h bk01 = ldB16(kb + hn * 72 + 32, g);
    v16h bk10 = ldB16(kb + (16 + hn) * 72, g);
    v16h bk11 = ldB16(kb + (16 + hn) * 72 + 32, g);

    v8f s0 = vzero, s1 = vzero;
    s0 = wmma16(aq0, bk00, s0);
    s0 = wmma16(aq1, bk01, s0);
    s1 = wmma16(aq0, bk10, s1);
    s1 = wmma16(aq1, bk11, s1);

    const float uc0 = Uw[tk0], uc1 = Uw[tk1];

    float l0[8], l1[8], tmax[8];
    #pragma unroll
    for (int r = 0; r < 8; ++r) {
      float a = s0[r] * 0.125f + ((ur[r] * uc0 < 1.0f) ? NEGV : 0.0f);
      float b = s1[r] * 0.125f + ((ur[r] * uc1 < 1.0f) ? NEGV : 0.0f);
      l0[r] = a; l1[r] = b;
      tmax[r] = fmaxf(a, b);
    }
    #pragma unroll
    for (int off = 1; off < 16; off <<= 1) {
      #pragma unroll
      for (int r = 0; r < 8; ++r)
        tmax[r] = fmaxf(tmax[r], __shfl_xor(tmax[r], off, 32));
    }
    float scale[8], psum[8];
    #pragma unroll
    for (int r = 0; r < 8; ++r) {
      float mn = fmaxf(rm[r], tmax[r]);
      scale[r] = __expf(rm[r] - mn);
      rm[r] = mn;
      l0[r] = __expf(l0[r] - mn);
      l1[r] = __expf(l1[r] - mn);
      psum[r] = l0[r] + l1[r];
    }
    #pragma unroll
    for (int off = 1; off < 16; off <<= 1) {
      #pragma unroll
      for (int r = 0; r < 8; ++r) psum[r] += __shfl_xor(psum[r], off, 32);
    }
    #pragma unroll
    for (int r = 0; r < 8; ++r) {
      rl[r] = rl[r] * scale[r] + psum[r];
      accO[0][r] *= scale[r];
      accO[1][r] *= scale[r];
      accO[2][r] *= scale[r];
      accO[3][r] *= scale[r];
    }

    // Restage P (16x32) through LDS into A-fragment layout (wave-private region)
    #pragma unroll
    for (int r = 0; r < 8; ++r) {
      int m = r + 8 * g;
      pw[m * 40 + hn] = (_Float16)l0[r];
      pw[m * 40 + 16 + hn] = (_Float16)l1[r];
    }
    __syncthreads();
    v16h ap = ldA16(pw + hn * 40, g);
    #pragma unroll
    for (int d = 0; d < 4; ++d) {
      int dim = d * 16 + hn;
      v16h bv = ldB16(vb + dim * 40, g);
      accO[d] = wmma16(ap, bv, accO[d]);
    }
  }

  // Normalize and restage O for the fused output projection
  float rinv[8];
  #pragma unroll
  for (int r = 0; r < 8; ++r) rinv[r] = 1.0f / rl[r];
  _Float16* ow = &o_lds[wave][0];
  __syncthreads();
  #pragma unroll
  for (int d = 0; d < 4; ++d) {
    #pragma unroll
    for (int r = 0; r < 8; ++r)
      ow[(r + 8 * g) * 72 + d * 16 + hn] = (_Float16)(accO[d][r] * rinv[r]);
  }
  __syncthreads();
  v16h ao0 = ldA16(ow + hn * 72, g);
  v16h ao1 = ldA16(ow + hn * 72 + 32, g);

  #pragma unroll
  for (int d = 0; d < 4; ++d) {
    int o = d * 16 + hn;
    v16h bw0 = ldB16(WcH + o * 64, g);
    v16h bw1 = ldB16(WcH + o * 64 + 32, g);
    float bb = bc[o];
    v8f acc = {bb, bb, bb, bb, bb, bb, bb, bb};
    acc = wmma16(ao0, bw0, acc);
    acc = wmma16(ao1, bw1, acc);
    #pragma unroll
    for (int r = 0; r < 8; ++r) {
      int t = r0 + r + 8 * g;
      int hh = (win >> 2) * SSW + t / SSW;
      int ww = (win & 3) * SSW + t % SSW;
      size_t gi = (size_t)o * HW2 + hh * HHW + ww;
      refined[gi] = x[gi] + acc[r];  // residual
    }
  }
}

// ---------------- conv3x3 + ReLU -> r3 (d_out region 0) ----------------
// grid (144, 4): blockIdx.y = group of 16 out-channels, blockIdx.x = 256-pixel tile.
__global__ __launch_bounds__(256) void conv3_kernel(
    const float* __restrict__ refined, const float* __restrict__ w3,
    const float* __restrict__ b3, float* __restrict__ r3out) {
  __shared__ float wsm[16 * 64 * 9];  // 36 KB
  const int cog = blockIdx.y;
  const int tid = threadIdx.x;
  for (int i = tid; i < 16 * 64 * 9; i += 256)
    wsm[i] = w3[(size_t)(cog * 16) * 576 + i];  // [j][ci][tap] contiguous
  __syncthreads();

  const int p = blockIdx.x * 256 + tid;
  const int y = p / HHW, xx = p % HHW;
  float acc[16];
  #pragma unroll
  for (int j = 0; j < 16; ++j) acc[j] = b3[cog * 16 + j];

  #pragma unroll
  for (int tap = 0; tap < 9; ++tap) {
    int yy = y + tap / 3 - 1;
    int xq = xx + tap % 3 - 1;
    if (yy >= 0 && yy < HHW && xq >= 0 && xq < HHW) {
      const float* rp = refined + yy * HHW + xq;
      for (int ci = 0; ci < 64; ++ci) {
        float v = rp[(size_t)ci * HW2];
        #pragma unroll
        for (int j = 0; j < 16; ++j)
          acc[j] = fmaf(v, wsm[j * 576 + ci * 9 + tap], acc[j]);
      }
    }
  }
  #pragma unroll
  for (int j = 0; j < 16; ++j)
    r3out[(size_t)(cog * 16 + j) * HW2 + p] = fmaxf(acc[j], 0.0f);
}

// ---------------- 1x1 conv + partition map ----------------
__global__ __launch_bounds__(256) void tail_kernel(
    const float* __restrict__ r3, const float* __restrict__ w4,
    const float* __restrict__ b4, float* __restrict__ out1,
    float* __restrict__ pmap) {
  int p = blockIdx.x * 256 + threadIdx.x;
  float acc = b4[0];
  for (int c = 0; c < 64; ++c) acc = fmaf(r3[(size_t)c * HW2 + p], w4[c], acc);
  out1[p] = acc;
  int y = p / HHW, x = p % HHW;
  int ym = y % SSW, xm = x % SSW;
  pmap[p] = (ym == 0 || ym == SSW - 1 || xm == 0 || xm == SSW - 1) ? 0.6f : 1.0f;
}

// ---------------- launch ----------------

extern "C" void kernel_launch(void* const* d_in, const int* in_sizes, int n_in,
                              void* d_out, int out_size, void* d_ws,
                              size_t ws_size, hipStream_t stream) {
  (void)in_sizes; (void)n_in; (void)out_size; (void)ws_size;
  const float* x   = (const float*)d_in[0];
  const float* ref = (const float*)d_in[1];
  const float* unc = (const float*)d_in[2];
  const float* ipw = (const float*)d_in[3];
  const float* ipb = (const float*)d_in[4];
  const float* opw = (const float*)d_in[5];
  const float* opb = (const float*)d_in[6];
  const float* w1  = (const float*)d_in[7];
  const float* b1  = (const float*)d_in[8];
  const float* w3  = (const float*)d_in[9];
  const float* b3  = (const float*)d_in[10];
  const float* w4  = (const float*)d_in[11];
  const float* b4  = (const float*)d_in[12];

  char* ws = (char*)d_ws;
  size_t off = 0;
  auto alloc = [&](size_t bytes) -> void* {
    void* p = ws + off;
    off = (off + bytes + 255) & ~(size_t)255;
    return p;
  };
  _Float16* Qh  = (_Float16*)alloc((size_t)NWIN * NN * DD * 2);
  _Float16* Kh  = (_Float16*)alloc((size_t)NWIN * NN * DD * 2);
  _Float16* Vt  = (_Float16*)alloc((size_t)NWIN * NN * DD * 2);
  float*    Ub  = (float*)   alloc((size_t)NWIN * NN * 4);
  _Float16* WqH = (_Float16*)alloc(4096 * 2);
  _Float16* WkH = (_Float16*)alloc(4096 * 2);
  _Float16* WvH = (_Float16*)alloc(4096 * 2);
  _Float16* WcH = (_Float16*)alloc(4096 * 2);
  float*    bc  = (float*)   alloc(64 * 4);
  float*    refined = (float*)alloc((size_t)DD * HW2 * 4);

  float* outp = (float*)d_out;
  float* r3   = outp;                     // 64*36864
  float* o1   = outp + (size_t)64 * HW2;  // 36864
  float* pm   = o1 + HW2;                 // 36864

  prep_w_kernel<<<16, 256, 0, stream>>>(ipw, opw, opb, w1, b1, WqH, WkH, WvH,
                                        WcH, bc);
  prep_u_kernel<<<144, 256, 0, stream>>>(unc, Ub);
  qkv_proj_kernel<<<dim3(36, 16), 256, 0, stream>>>(x, ref, ipb, WqH, WkH, WvH,
                                                    Qh, Kh, Vt);
  flash_kernel<<<dim3(36, 16), 128, 0, stream>>>(Qh, Kh, Vt, Ub, WcH, bc, x,
                                                 refined);
  conv3_kernel<<<dim3(144, 4), 256, 0, stream>>>(refined, w3, b3, r3);
  tail_kernel<<<144, 256, 0, stream>>>(r3, w4, b4, o1, pm);
}
